// MoE_11673721110901
// MI455X (gfx1250) — matrix operand order repaired and checked by reading, not compiled, not run on hardware
//
#include <hip/hip_runtime.h>
#include <hip/hip_bf16.h>
#include <math.h>

// ---------------------------------------------------------------------------
// MoE top-1 (DeepSpeed-style) forward for MI455X / gfx1250, wave32 + WMMA.
// B=4 S=2048 H=1024 F=4096 E=8 -> T=8192, capacity C=1024.
// ---------------------------------------------------------------------------

typedef __attribute__((ext_vector_type(16))) _Float16 v16h;
typedef __attribute__((ext_vector_type(8)))  _Float16 v8h;
typedef __attribute__((ext_vector_type(2)))  _Float16 v2h;
typedef __attribute__((ext_vector_type(8)))  float    v8f;

#define CAT16(lo, hi) __builtin_shufflevector((lo), (hi), 0,1,2,3,4,5,6,7,8,9,10,11,12,13,14,15)

static constexpr int kT   = 8192;   // tokens
static constexpr int kH   = 1024;   // hidden
static constexpr int kF   = 4096;   // ffn
static constexpr int kE   = 8;      // experts
static constexpr int kCap = 1024;   // capacity = ceil(T/E)

// tile config
static constexpr int TM = 128, TN = 128, TK = 32;
static constexpr int LSTR = 40;     // LDS row stride in halves (padded: 80B, 16B aligned)

// ---------------------------------------------------------------------------
// 0) init: zero out[T*H], slot_token=-1, gates_sum=0
// ---------------------------------------------------------------------------
__global__ __launch_bounds__(256) void moe_init_kernel(float* __restrict__ out,
                                                       int* __restrict__ slot_token,
                                                       float* __restrict__ gates_sum) {
  const int gid = blockIdx.x * 256 + threadIdx.x;   // 8192 blocks -> 2,097,152 threads
  float4 z; z.x = 0.f; z.y = 0.f; z.z = 0.f; z.w = 0.f;
  ((float4*)out)[gid] = z;                          // covers exactly T*H = 8,388,608 floats
  if (gid < kE * kCap) slot_token[gid] = -1;
  if (gid < kE)        gates_sum[gid]  = 0.f;
}

// ---------------------------------------------------------------------------
// 1) gating: logits = x@wg, softmax, argmax. One wave (32 lanes) per token.
// ---------------------------------------------------------------------------
__global__ __launch_bounds__(256) void moe_gating_kernel(const float* __restrict__ x,
                                                         const float* __restrict__ wg,
                                                         int* __restrict__ tok_expert,
                                                         float* __restrict__ tok_gate,
                                                         float* __restrict__ gates_sum) {
  const int wave = threadIdx.x >> 5;
  const int lane = threadIdx.x & 31;
  const int t = blockIdx.x * 8 + wave;

  float acc[kE];
#pragma unroll
  for (int e = 0; e < kE; ++e) acc[e] = 0.f;

  const float* xrow = x + (size_t)t * kH;
  for (int h = lane; h < kH; h += 32) {
    const float xv = xrow[h];
    const float* wr = wg + (size_t)h * kE;
#pragma unroll
    for (int e = 0; e < kE; ++e) acc[e] += xv * wr[e];
  }
  // wave reduction (wave32)
#pragma unroll
  for (int e = 0; e < kE; ++e) {
#pragma unroll
    for (int off = 16; off > 0; off >>= 1) acc[e] += __shfl_xor(acc[e], off, 32);
  }
  // softmax + first-argmax (all lanes redundantly)
  float m = acc[0]; int idx = 0;
#pragma unroll
  for (int e = 1; e < kE; ++e) { if (acc[e] > m) { m = acc[e]; idx = e; } }
  float g[kE]; float s = 0.f;
#pragma unroll
  for (int e = 0; e < kE; ++e) { g[e] = __expf(acc[e] - m); s += g[e]; }
  const float inv = 1.0f / s;
  if (lane < kE) atomicAdd(&gates_sum[lane], g[lane] * inv);
  if (lane == 0) { tok_expert[t] = idx; tok_gate[t] = g[idx] * inv; }
}

// ---------------------------------------------------------------------------
// 2) scan: sequential per-expert prefix counters over token order (1 wave).
//    Emits slot_token[e][slot] for kept tokens, exp_counts + l_aux to tail.
// ---------------------------------------------------------------------------
__global__ __launch_bounds__(32) void moe_scan_kernel(const int* __restrict__ tok_expert,
                                                      const float* __restrict__ gates_sum,
                                                      int* __restrict__ slot_token,
                                                      float* __restrict__ out_tail) {
  const int lane = threadIdx.x;
  const unsigned long long below = (lane == 0) ? 0ull : ((1ull << lane) - 1ull);
  int base[kE];
#pragma unroll
  for (int e = 0; e < kE; ++e) base[e] = 0;

  for (int t0 = 0; t0 < kT; t0 += 32) {
    const int t = t0 + lane;
    const int e = tok_expert[t];
    int slot = -1;
#pragma unroll
    for (int ex = 0; ex < kE; ++ex) {
      const unsigned long long b = __ballot(e == ex);
      if (e == ex) slot = base[ex] + (int)__popcll(b & below);
      base[ex] += (int)__popcll(b);
    }
    if (slot >= 0 && slot < kCap) slot_token[e * kCap + slot] = t;
  }
  if (lane < kE) out_tail[1 + lane] = (float)base[lane];   // exp_counts (pre-drop)
  if (lane == 0) {
    float la = 0.f;
    const float invT = 1.0f / (float)kT;
#pragma unroll
    for (int e = 0; e < kE; ++e) la += (gates_sum[e] * invT) * ((float)base[e] * invT);
    out_tail[0] = la * (float)kE;                          // l_aux
  }
}

// ---------------------------------------------------------------------------
// 3/4) Expert GEMM, f16 WMMA with f32 accumulate.
//   L1: A = gather(x) [C,H] -> gelu(A@w1 + b1) stored f16 into h [E,C,F]
//       A staged via VALU f32->f16 cvt, vectorized ds_store_b128.
//   L2: A = h [C,F] (already f16) -> staged via ASYNC global->LDS copies
//       (global_load_async_to_lds_b128, ASYNCcnt), then
//       scatter gate*(A@w2 + b2) into out[token].
// Block = 256 threads (8 waves); tile 128x128x32; wave sub-tile 64x32.
// ---------------------------------------------------------------------------
template <bool L1>
__global__ __launch_bounds__(256) void moe_gemm_kernel(const float* __restrict__ x,
                                                       const _Float16* __restrict__ hin,
                                                       const float* __restrict__ W,
                                                       const float* __restrict__ bias,
                                                       const int* __restrict__ slot_token,
                                                       const float* __restrict__ tok_gate,
                                                       _Float16* __restrict__ hout,
                                                       float* __restrict__ out) {
  constexpr int Kdim = L1 ? kH : kF;
  constexpr int Ndim = L1 ? kF : kH;

  const int e     = blockIdx.z;
  const int cBase = blockIdx.y * TM;
  const int nBase = blockIdx.x * TN;
  const int tid   = threadIdx.x;
  const int lane  = tid & 31;
  const int waveM = (tid >> 5) >> 2;   // 0..1
  const int waveN = (tid >> 5) & 3;    // 0..3
  const int lhi   = lane >> 4;         // 0/1
  const int llo   = lane & 15;

  __shared__ __align__(16) _Float16 As[TM * LSTR];
  __shared__ __align__(16) _Float16 Bs[TN * LSTR];

  // A staging: 2 threads per row; 16 K-halves (32B) each.
  const int arow  = tid >> 1;
  const int akoff = (tid & 1) * 16;
  const float*    aSrcF = nullptr;
  const _Float16* aSrcH = nullptr;
  if (L1) {
    const int tokenA = slot_token[e * kCap + cBase + arow];
    if (tokenA >= 0) aSrcF = x + (size_t)tokenA * kH + akoff;
  } else {
    aSrcH = hin + ((size_t)e * kCap + cBase + arow) * (size_t)kF + akoff;
  }
  const unsigned aLds = (unsigned)(size_t)&As[arow * LSTR + akoff];  // LDS byte address

  // B staging: thread owns K rows {2kp, 2kp+1} x 8 N columns -> packed b32 LDS stores.
  const int kp = tid >> 4;             // 0..15 -> k = 2kp
  const int n0 = (tid & 15) * 8;
  const float* bSrc = W + ((size_t)e * Kdim + 2 * kp) * (size_t)Ndim + nBase + n0;

  v8f acc[4][2];
#pragma unroll
  for (int mi = 0; mi < 4; ++mi)
#pragma unroll
    for (int ni = 0; ni < 2; ++ni) { v8f z = {}; acc[mi][ni] = z; }

  for (int k0 = 0; k0 < Kdim; k0 += TK) {
    if (k0 + TK < Kdim)
      __builtin_prefetch(bSrc + (size_t)(k0 + TK) * Ndim, 0, 3);

    // ---- stage A tile ----
    if (L1) {
      // f32 -> f16 on the fly, vectorized LDS stores
      float4 va[4];
      if (aSrcF) {
        const float4* p = (const float4*)(aSrcF + k0);
#pragma unroll
        for (int j = 0; j < 4; ++j) va[j] = p[j];
      } else {
#pragma unroll
        for (int j = 0; j < 4; ++j) { va[j].x = 0.f; va[j].y = 0.f; va[j].z = 0.f; va[j].w = 0.f; }
      }
      v8h h0, h1;
      h0[0] = (_Float16)va[0].x; h0[1] = (_Float16)va[0].y;
      h0[2] = (_Float16)va[0].z; h0[3] = (_Float16)va[0].w;
      h0[4] = (_Float16)va[1].x; h0[5] = (_Float16)va[1].y;
      h0[6] = (_Float16)va[1].z; h0[7] = (_Float16)va[1].w;
      h1[0] = (_Float16)va[2].x; h1[1] = (_Float16)va[2].y;
      h1[2] = (_Float16)va[2].z; h1[3] = (_Float16)va[2].w;
      h1[4] = (_Float16)va[3].x; h1[5] = (_Float16)va[3].y;
      h1[6] = (_Float16)va[3].z; h1[7] = (_Float16)va[3].w;
      *(v8h*)&As[arow * LSTR + akoff]     = h0;
      *(v8h*)&As[arow * LSTR + akoff + 8] = h1;
    } else {
      // CDNA5 async global->LDS copy (ASYNCcnt); IOFFSET applies to both addrs.
      const unsigned long long ga = (unsigned long long)(size_t)(aSrcH + k0);
      asm volatile(
          "global_load_async_to_lds_b128 %0, %1, off\n\t"
          "global_load_async_to_lds_b128 %0, %1, off offset:16"
          :
          : "v"(aLds), "v"(ga)
          : "memory");
    }
    // ---- stage B tile transposed: LDS layout [N][K], packed (k,k+1) pairs ----
    {
      const float* p0 = bSrc + (size_t)k0 * Ndim;
      const float* p1 = p0 + Ndim;
      float4 r0a = ((const float4*)p0)[0], r0b = ((const float4*)p0)[1];
      float4 r1a = ((const float4*)p1)[0], r1b = ((const float4*)p1)[1];
      const float lo[8] = {r0a.x, r0a.y, r0a.z, r0a.w, r0b.x, r0b.y, r0b.z, r0b.w};
      const float hi[8] = {r1a.x, r1a.y, r1a.z, r1a.w, r1b.x, r1b.y, r1b.z, r1b.w};
#pragma unroll
      for (int j = 0; j < 8; ++j) {
        v2h pk; pk[0] = (_Float16)lo[j]; pk[1] = (_Float16)hi[j];
        *(v2h*)&Bs[(n0 + j) * LSTR + 2 * kp] = pk;
      }
    }
    if (!L1) asm volatile("s_wait_asynccnt 0" ::: "memory");
    __syncthreads();

    // ---- fragments (layouts per CDNA5 ISA §7.12.2) ----
    v16h afrag[4], bfrag[2];
#pragma unroll
    for (int mi = 0; mi < 4; ++mi) {
      const _Float16* ab = &As[(waveM * 64 + mi * 16 + llo) * LSTR];
      const v8h lo = *(const v8h*)(ab + lhi * 8);        // K = lhi*8 .. +8
      const v8h hi = *(const v8h*)(ab + 16 + lhi * 8);   // K = 16+lhi*8 .. +8
      afrag[mi] = CAT16(lo, hi);
    }
#pragma unroll
    for (int ni = 0; ni < 2; ++ni) {
      const _Float16* bb = &Bs[(waveN * 32 + ni * 16 + llo) * LSTR];
      const v8h lo = *(const v8h*)(bb + lhi * 8);        // K = lhi*8 .. +8
      const v8h hi = *(const v8h*)(bb + 16 + lhi * 8);   // K = 16+lhi*8 .. +8
      bfrag[ni] = CAT16(lo, hi);
    }
#pragma unroll
    for (int mi = 0; mi < 4; ++mi)
#pragma unroll
      for (int ni = 0; ni < 2; ++ni)
        acc[mi][ni] = __builtin_amdgcn_wmma_f32_16x16x32_f16(
            false, afrag[mi], false, bfrag[ni], (short)0, acc[mi][ni], false, false);
    __syncthreads();
  }

  // ---- epilogue ----
#pragma unroll
  for (int mi = 0; mi < 4; ++mi) {
#pragma unroll
    for (int ni = 0; ni < 2; ++ni) {
      const int col  = nBase + waveN * 32 + ni * 16 + llo;
      const int rloc = waveM * 64 + mi * 16 + lhi * 8;
      const float bv = bias[e * Ndim + col];
#pragma unroll
      for (int v = 0; v < 8; ++v) {
        const int row = cBase + rloc + v;
        const float val = acc[mi][ni][v] + bv;
        if (L1) {
          // tanh-approx gelu (jax.nn.gelu default)
          const float gel = 0.5f * val *
              (1.0f + tanhf(0.7978845608028654f * (val + 0.044715f * val * val * val)));
          hout[((size_t)e * kCap + row) * (size_t)kF + col] = (_Float16)gel;
        } else {
          const int tk = slot_token[e * kCap + row];
          if (tk >= 0) out[(size_t)tk * kH + col] = tok_gate[tk] * val;
        }
      }
    }
  }
}

// ---------------------------------------------------------------------------
// launch
// ---------------------------------------------------------------------------
extern "C" void kernel_launch(void* const* d_in, const int* in_sizes, int n_in,
                              void* d_out, int out_size, void* d_ws, size_t ws_size,
                              hipStream_t stream) {
  const float* x  = (const float*)d_in[0];   // [T,H]
  const float* wg = (const float*)d_in[1];   // [H,E]
  const float* w1 = (const float*)d_in[2];   // [E,H,F]
  const float* b1 = (const float*)d_in[3];   // [E,F]
  const float* w2 = (const float*)d_in[4];   // [E,F,H]
  const float* b2 = (const float*)d_in[5];   // [E,H]

  float* out      = (float*)d_out;           // [T*H] then l_aux then exp_counts[8]
  float* out_tail = out + (size_t)kT * kH;

  char* ws = (char*)d_ws;
  float*     gates_sum  = (float*)(ws + 0);                 // 8 f32
  int*       tok_expert = (int*)(ws + 256);                 // T ints
  float*     tok_gate   = (float*)(ws + 256 + 32768);       // T f32
  int*       slot_token = (int*)(ws + 256 + 65536);         // E*Cap ints
  _Float16*  hbuf       = (_Float16*)(ws + 98560);          // E*Cap*F halves (64 MB)

  // 0) init
  moe_init_kernel<<<dim3((kT * kH) / (256 * 4)), dim3(256), 0, stream>>>(out, slot_token, gates_sum);
  // 1) gating
  moe_gating_kernel<<<dim3(kT / 8), dim3(256), 0, stream>>>(x, wg, tok_expert, tok_gate, gates_sum);
  // 2) sequential scan (order-dependent capacity assignment)
  moe_scan_kernel<<<dim3(1), dim3(32), 0, stream>>>(tok_expert, gates_sum, slot_token, out_tail);
  // 3) GEMM1: gather + gelu -> h
  moe_gemm_kernel<true><<<dim3(kF / TN, kCap / TM, kE), dim3(256), 0, stream>>>(
      x, nullptr, w1, b1, slot_token, tok_gate, hbuf, nullptr);
  // 4) GEMM2: h @ w2 + b2, gate-scaled scatter into out (async-staged A)
  moe_gemm_kernel<false><<<dim3(kH / TN, kCap / TM, kE), dim3(256), 0, stream>>>(
      nullptr, hbuf, w2, b2, slot_token, tok_gate, nullptr, out);
}
